// semantic_GCN_21534966022326
// MI455X (gfx1250) — compile-verified, matrix operand order checked by reading, not run on hardware
//
#include <hip/hip_runtime.h>

#define HID 256
#define NNODES 100000
#define NEDGES 300000
#define NT0 40000
#define NT1 30000
#define NT2 30000
#define DT0 512
#define DT1 256
#define DT2 128

typedef __attribute__((ext_vector_type(16))) __bf16 v16bf;
typedef __attribute__((ext_vector_type(8)))  __bf16 v8bf;
typedef __attribute__((ext_vector_type(8)))  float  v8f;

union BF16x16 { v16bf v; v8bf h[2]; };

static inline int cdiv(long a, long b) { return (int)((a + b - 1) / b); }

// ---------------- elementwise helpers ----------------

__global__ void fill_f32_kernel(float* __restrict__ p, float v, long n) {
    long i = (long)blockIdx.x * blockDim.x + threadIdx.x;
    if (i < n) p[i] = v;
}

__global__ void cvt_bf16_kernel(const float* __restrict__ x, __bf16* __restrict__ y, long n) {
    long i = (long)blockIdx.x * blockDim.x + threadIdx.x;
    if (i < n) y[i] = (__bf16)x[i];
}

// gc2_weight is [in=256, out=256] row-major with y = x @ W.
// GEMM kernel wants Wr[out, in] row-major (y = x @ Wr.T), so transpose here.
__global__ void cvt_bf16_t256_kernel(const float* __restrict__ w, __bf16* __restrict__ y) {
    int idx = blockIdx.x * blockDim.x + threadIdx.x;   // idx = i*256 + o
    if (idx < HID * HID) {
        int i = idx / HID;
        int o = idx % HID;
        y[(long)o * HID + i] = (__bf16)w[idx];
    }
}

__global__ void degree_kernel(const int* __restrict__ src, const int* __restrict__ dst,
                              float* __restrict__ degO, float* __restrict__ degI, int e_count) {
    int e = blockIdx.x * blockDim.x + threadIdx.x;
    if (e < e_count) {
        atomicAdd(&degO[src[e]], 1.0f);
        atomicAdd(&degI[dst[e]], 1.0f);
    }
}

__global__ void norm_kernel(float* __restrict__ d, int n) {
    int i = blockIdx.x * blockDim.x + threadIdx.x;
    if (i < n) d[i] = rsqrtf(fmaxf(d[i], 1.0f));
}

// msg = (h * norm_out)[src]; agg[dst] += msg   (one edge per block, one feature per lane)
__global__ void scatter_kernel(const float* __restrict__ h, const float* __restrict__ normO,
                               const int* __restrict__ src, const int* __restrict__ dst,
                               float* __restrict__ agg) {
    int e = blockIdx.x;
    int f = threadIdx.x;
    int s = src[e];
    int d = dst[e];
    float v = h[(long)s * HID + f] * normO[s];
    atomicAdd(&agg[(long)d * HID + f], v);
}

// layer-1 epilogue: h = relu(agg * norm_in + gc1_bias)   (no weight matrix)
__global__ void post1_kernel(const float* __restrict__ agg, const float* __restrict__ normI,
                             const float* __restrict__ bias, float* __restrict__ h, long total) {
    long i = (long)blockIdx.x * blockDim.x + threadIdx.x;
    if (i < total) {
        int node = (int)(i / HID);
        int f = (int)(i % HID);
        h[i] = fmaxf(agg[i] * normI[node] + bias[f], 0.0f);
    }
}

// layer-2 pre-GEMM: hb = bf16(agg * norm_in)
__global__ void pre2_kernel(const float* __restrict__ agg, const float* __restrict__ normI,
                            __bf16* __restrict__ hb, long total) {
    long i = (long)blockIdx.x * blockDim.x + threadIdx.x;
    if (i < total) {
        int node = (int)(i / HID);
        hb[i] = (__bf16)(agg[i] * normI[node]);
    }
}

__global__ void copy_sem_kernel(const float* __restrict__ sw, float* __restrict__ out, int n) {
    int i = threadIdx.x;
    if (i < n) out[i] = sw[i];
}

// ---------------- WMMA GEMM ----------------
// Y[N, 256] = (relu?)(X[N, K] @ Wr[256, K].T + bias[256])
// One wave32 per 16x16 output tile; K stepped by 32 via v_wmma_f32_16x16x32_bf16.
// VGPR layouts per CDNA5 ISA 7.12.2:
//   A: lane L holds row M = L&15; elems [0..7] = K(half*8 .. half*8+7),
//      elems [8..15] = K(16+half*8 ..), half = L>>4. B mirrors A over columns.
//   C/D: lane L holds N = L&15; VGPR v holds M = v + 8*half.
__global__ void gemm_bf16_kernel(const __bf16* __restrict__ X,
                                 const __bf16* __restrict__ Wr,
                                 const float* __restrict__ bias,
                                 float* __restrict__ Y,
                                 int nrows, int K, int doRelu) {
    const int lane   = threadIdx.x & 31;
    const int wave   = threadIdx.x >> 5;
    const int waveId = blockIdx.x * (blockDim.x >> 5) + wave;
    const int nColTiles = HID / 16;                 // 16
    const int tileRow = waveId / nColTiles;
    const int tileCol = waveId % nColTiles;
    if (tileRow * 16 >= nrows) return;              // wave-uniform: EXEC stays all-1s

    const int half = lane >> 4;
    const int r    = lane & 15;
    const int rowA = tileRow * 16 + r;
    const int colB = tileCol * 16 + r;

    // bias is per output column (N = r for every acc VGPR) -> fold into accumulator init
    const float bval = bias[tileCol * 16 + r];
    v8f acc;
#pragma unroll
    for (int v = 0; v < 8; ++v) acc[v] = bval;

    const __bf16* aptr = X  + (long)rowA * K;
    const __bf16* bptr = Wr + (long)colB * K;

    for (int k = 0; k < K; k += 32) {
        BF16x16 a, b;
        a.h[0] = *(const v8bf*)(aptr + k +      half * 8);
        a.h[1] = *(const v8bf*)(aptr + k + 16 + half * 8);
        b.h[0] = *(const v8bf*)(bptr + k +      half * 8);
        b.h[1] = *(const v8bf*)(bptr + k + 16 + half * 8);
        acc = __builtin_amdgcn_wmma_f32_16x16x32_bf16(
            /*neg_a=*/false, a.v, /*neg_b=*/false, b.v,
            /*c_mod=*/(short)0, acc, /*reuse_a=*/false, /*reuse_b=*/false);
    }

    float* yptr = Y + (long)(tileRow * 16 + half * 8) * HID + tileCol * 16 + r;
#pragma unroll
    for (int v = 0; v < 8; ++v) {
        float val = acc[v];
        if (doRelu) val = fmaxf(val, 0.0f);
        yptr[(long)v * HID] = val;                  // row M = v + 8*half
    }
}

// ---------------- host launcher ----------------

extern "C" void kernel_launch(void* const* d_in, const int* in_sizes, int n_in,
                              void* d_out, int out_size, void* d_ws, size_t ws_size,
                              hipStream_t stream) {
    const float* feat0   = (const float*)d_in[0];
    const float* feat1   = (const float*)d_in[1];
    const float* feat2   = (const float*)d_in[2];
    const float* fc_w0   = (const float*)d_in[3];
    const float* fc_b0   = (const float*)d_in[4];
    const float* fc_w1   = (const float*)d_in[5];
    const float* fc_b1   = (const float*)d_in[6];
    const float* fc_w2   = (const float*)d_in[7];
    const float* fc_b2   = (const float*)d_in[8];
    const float* gc1_b   = (const float*)d_in[9];
    const float* gc2_w   = (const float*)d_in[10];
    const float* gc2_b   = (const float*)d_in[11];
    const float* sem_w   = (const float*)d_in[12];
    const int*   src     = (const int*)d_in[13];
    const int*   dst     = (const int*)d_in[14];
    float*       out     = (float*)d_out;

    // workspace carve-out
    char* ws = (char*)d_ws;
    auto carve = [&](size_t bytes) -> char* {
        char* p = ws;
        ws += (bytes + 255) & ~(size_t)255;
        return p;
    };
    float*  normO = (float*)carve((size_t)NNODES * 4);
    float*  normI = (float*)carve((size_t)NNODES * 4);
    float*  h     = (float*)carve((size_t)NNODES * HID * 4);
    float*  agg   = (float*)carve((size_t)NNODES * HID * 4);
    __bf16* hb    = (__bf16*)carve((size_t)NNODES * HID * 2);
    __bf16* xb    = (__bf16*)carve((size_t)NT0 * DT0 * 2);   // largest feature, reused per type
    __bf16* wb    = (__bf16*)carve((size_t)HID * DT0 * 2);   // largest weight, reused

    const long total = (long)NNODES * HID;

    // --- degrees -> rsqrt norms ---
    fill_f32_kernel<<<cdiv(NNODES, 256), 256, 0, stream>>>(normO, 0.0f, NNODES);
    fill_f32_kernel<<<cdiv(NNODES, 256), 256, 0, stream>>>(normI, 0.0f, NNODES);
    degree_kernel<<<cdiv(NEDGES, 256), 256, 0, stream>>>(src, dst, normO, normI, NEDGES);
    norm_kernel<<<cdiv(NNODES, 256), 256, 0, stream>>>(normO, NNODES);
    norm_kernel<<<cdiv(NNODES, 256), 256, 0, stream>>>(normI, NNODES);

    // --- per-type projections into h (bias folded, no relu) ---
    struct Proj { const float* x; const float* w; const float* b; int n; int k; long off; };
    const Proj proj[3] = {
        { feat0, fc_w0, fc_b0, NT0, DT0, 0 },
        { feat1, fc_w1, fc_b1, NT1, DT1, (long)NT0 * HID },
        { feat2, fc_w2, fc_b2, NT2, DT2, (long)(NT0 + NT1) * HID },
    };
    for (int t = 0; t < 3; ++t) {
        long nx = (long)proj[t].n * proj[t].k;
        long nw = (long)HID * proj[t].k;
        cvt_bf16_kernel<<<cdiv(nx, 256), 256, 0, stream>>>(proj[t].x, xb, nx);
        cvt_bf16_kernel<<<cdiv(nw, 256), 256, 0, stream>>>(proj[t].w, wb, nw);
        int tiles = (proj[t].n / 16) * (HID / 16);
        gemm_bf16_kernel<<<cdiv(tiles, 8), 256, 0, stream>>>(
            xb, wb, proj[t].b, h + proj[t].off, proj[t].n, proj[t].k, /*relu=*/0);
    }

    // --- GraphConv layer 1 (weight=False) ---
    fill_f32_kernel<<<cdiv(total, 256), 256, 0, stream>>>(agg, 0.0f, total);
    scatter_kernel<<<NEDGES, HID, 0, stream>>>(h, normO, src, dst, agg);
    post1_kernel<<<cdiv(total, 256), 256, 0, stream>>>(agg, normI, gc1_b, h, total);

    // --- GraphConv layer 2 (weight + bias + relu) ---
    fill_f32_kernel<<<cdiv(total, 256), 256, 0, stream>>>(agg, 0.0f, total);
    scatter_kernel<<<NEDGES, HID, 0, stream>>>(h, normO, src, dst, agg);
    pre2_kernel<<<cdiv(total, 256), 256, 0, stream>>>(agg, normI, hb, total);
    cvt_bf16_t256_kernel<<<cdiv(HID * HID, 256), 256, 0, stream>>>(gc2_w, wb);
    {
        int tiles = (NNODES / 16) * (HID / 16);
        gemm_bf16_kernel<<<cdiv(tiles, 8), 256, 0, stream>>>(
            hb, wb, gc2_b, out, NNODES, HID, /*relu=*/1);
    }

    // --- append semantic_weight (5 floats) ---
    copy_sem_kernel<<<1, 32, 0, stream>>>(sem_w, out + total, 5);
}
